// AttentionWrapper_78666620993915
// MI455X (gfx1250) — compile-verified, matrix operand order checked by reading, not run on hardware
//
#include <hip/hip_runtime.h>
#include <hip/hip_bf16.h>
#include <stdint.h>

// ---------------------------------------------------------------------------
// Problem constants (B=4, N=2048, C=768, H=12, D=64)
// ---------------------------------------------------------------------------
#define B_   4
#define N_   2048
#define C_   768
#define H_   12
#define D_   64
#define M_   (B_ * N_)     // 8192 rows
#define OC3  (3 * C_)      // 2304

typedef __attribute__((ext_vector_type(16))) __bf16        v16bf;
typedef __attribute__((ext_vector_type(8)))  float         v8f;
typedef __attribute__((ext_vector_type(4)))  unsigned int  uv4;
typedef __attribute__((ext_vector_type(4)))  unsigned int  u32x4;
typedef __attribute__((ext_vector_type(4)))  int           i32x4;
typedef __attribute__((ext_vector_type(8)))  int           i32x8;

union Frag {
    v16bf           bf;
    uv4             q4[2];
    unsigned short  us[16];
};

__device__ __forceinline__ unsigned short f32_to_bf16(float f) {
    unsigned int u = __builtin_bit_cast(unsigned int, f);
    u += 0x7FFFu + ((u >> 16) & 1u);          // round to nearest even
    return (unsigned short)(u >> 16);
}

__device__ __forceinline__ v8f wmma_bf16(const Frag& a, const Frag& b, v8f c) {
    return __builtin_amdgcn_wmma_f32_16x16x32_bf16(
        /*neg_a=*/false, a.bf, /*neg_b=*/false, b.bf,
        /*c_mod=*/(short)0, c, /*reuse_a=*/false, /*reuse_b=*/false);
}

#if defined(__gfx1250__) && __has_builtin(__builtin_amdgcn_tensor_load_to_lds) && \
    __has_builtin(__builtin_amdgcn_s_wait_tensorcnt)
#define USE_TDM 1
#else
#define USE_TDM 0
#endif

// ---------------------------------------------------------------------------
// fp32 -> bf16 conversion (grid-stride)
// ---------------------------------------------------------------------------
__global__ __launch_bounds__(256) void cvt_f32_bf16_kernel(
    const float* __restrict__ src, unsigned short* __restrict__ dst, int n)
{
    int i = blockIdx.x * blockDim.x + threadIdx.x;
    int stride = gridDim.x * blockDim.x;
    for (; i < n; i += stride) dst[i] = f32_to_bf16(src[i]);
}

// ---------------------------------------------------------------------------
// QKV GEMM: [8192,768] x [2304,768]^T + bias -> q/k/v bf16 [B,H,N,D]
// Block: 256 thr (8 waves), tile 128(M) x 128(O). Wave = 32(M) x 64(O).
// ---------------------------------------------------------------------------
__global__ __launch_bounds__(256) void qkv_gemm_kernel(
    const unsigned short* __restrict__ xb,   // [8192,768] bf16
    const unsigned short* __restrict__ wb,   // [2304,768] bf16
    const float*          __restrict__ bias, // [2304] f32
    unsigned short* __restrict__ qo,
    unsigned short* __restrict__ ko,
    unsigned short* __restrict__ vo)
{
    const int wave  = threadIdx.x >> 5;
    const int lane  = threadIdx.x & 31;
    const int ln    = lane & 15;
    const int hl    = lane >> 4;
    const int Mbase = blockIdx.y * 128 + (wave & 3) * 32;
    const int Obase = blockIdx.x * 128 + (wave >> 2) * 64;

    v8f acc[2][4];
#pragma unroll
    for (int i = 0; i < 2; i++)
#pragma unroll
        for (int j = 0; j < 4; j++)
#pragma unroll
            for (int r = 0; r < 8; r++) acc[i][j][r] = 0.0f;

#pragma unroll 2
    for (int k0 = 0; k0 < C_; k0 += 32) {
        Frag a[2];
#pragma unroll
        for (int mm = 0; mm < 2; mm++) {
            const unsigned short* p =
                xb + (size_t)(Mbase + mm * 16 + ln) * C_ + k0 + 8 * hl;
            a[mm].q4[0] = *(const uv4*)p;
            a[mm].q4[1] = *(const uv4*)(p + 16);
        }
#pragma unroll
        for (int nn = 0; nn < 4; nn++) {
            Frag bf;
            const unsigned short* p =
                wb + (size_t)(Obase + nn * 16 + ln) * C_ + k0 + 8 * hl;
            bf.q4[0] = *(const uv4*)p;
            bf.q4[1] = *(const uv4*)(p + 16);
#pragma unroll
            for (int mm = 0; mm < 2; mm++)
                acc[mm][nn] = wmma_bf16(a[mm], bf, acc[mm][nn]);
        }
    }

    // Epilogue: bias + scatter into q/k/v [B,H,N,D] (bf16)
#pragma unroll
    for (int nn = 0; nn < 4; nn++) {
        const int o  = Obase + nn * 16 + ln;
        const float bv = bias[o];
        const int which = o / C_;
        const int rem   = o % C_;
        const int hh    = rem >> 6;
        const int d     = rem & 63;
        unsigned short* dst = (which == 0) ? qo : (which == 1) ? ko : vo;
#pragma unroll
        for (int mm = 0; mm < 2; mm++) {
            const int mrow = Mbase + mm * 16 + 8 * hl;
#pragma unroll
            for (int r = 0; r < 8; r++) {
                const int m  = mrow + r;
                const int b  = m >> 11;     // / 2048
                const int n  = m & 2047;
                const size_t off =
                    ((size_t)(b * H_ + hh) * N_ + n) * D_ + d;
                dst[off] = f32_to_bf16(acc[mm][nn][r] + bv);
            }
        }
    }
}

// ---------------------------------------------------------------------------
// Flash attention: per (b,h), 128 q rows per block (8 waves x 16 rows),
// streaming 32-key tiles through double-buffered LDS.
// Sᵀ = K·Qᵀ, Oᵀ += Vᵀ·P (all WMMA bf16). K tile staged by the Tensor Data
// Mover (async DMA overlapped with the previous tile's WMMA work).
// ---------------------------------------------------------------------------
#define KT      32
#define NT      (N_ / KT)   // 64 key tiles
#define KPITCH  72   // 144B rows = 128B data + 16B TDM pad (16B multiple)
#define VPITCH  40   // 80B rows  (16B multiple)

__global__ __launch_bounds__(256) void flash_attn_kernel(
    const unsigned short* __restrict__ q,   // [B,H,N,D] bf16
    const unsigned short* __restrict__ k,
    const unsigned short* __restrict__ v,
    unsigned short* __restrict__ o)         // [B,N,H*D] bf16
{
    __shared__ __align__(16) unsigned short Ksh[2][KT * KPITCH];
    __shared__ __align__(16) unsigned short VTsh[2][D_ * VPITCH];

    const int wave = threadIdx.x >> 5;
    const int lane = threadIdx.x & 31;
    const int ln   = lane & 15;
    const int hl   = lane >> 4;
    const int b    = blockIdx.z;
    const int h    = blockIdx.y;
    const int qrow = blockIdx.x * 128 + wave * 16;   // wave's 16 q rows

    const size_t bh = ((size_t)b * H_ + h) * N_;
    const unsigned short* qp = q + bh * D_;
    const unsigned short* kp = k + bh * D_;
    const unsigned short* vp = v + bh * D_;

    // Per-thread cooperative-staging coordinates
    const int kr = threadIdx.x >> 3;        // 0..31 key row
    const int kc = (threadIdx.x & 7) * 8;   // 0..56 d col (8 elems)

    // Q B-fragments for this wave's 16 rows (held for whole loop)
    Frag qf[2];
#pragma unroll
    for (int db = 0; db < 2; db++) {
        const unsigned short* p =
            qp + (size_t)(qrow + ln) * D_ + db * 32 + 8 * hl;
        qf[db].q4[0] = *(const uv4*)p;
        qf[db].q4[1] = *(const uv4*)(p + 16);
    }

#if USE_TDM
    // Constant part of the Tensor-DMA descriptor (ISA §8.4, D# group 1):
    //  data_size=2B; pad_enable; pad_interval code 4 (=32 DWORDs = 128B row);
    //  pad_amount code 3 (=4 DWORDs = 16B) -> LDS pitch 144B = KPITCH.
    //  tensor_dim0 = 64 elems, tensor_dim1 = 32 rows (== tile dims),
    //  tile_dim0 = 64, tile_dim1 = 32, stride0 = 64 elems.
    i32x8 g1;
    g1[0] = (1 << 16) | (1 << 20) | (4 << 22) | (3 << 25);
    g1[1] = (int)(((unsigned)D_ & 0xFFFFu) << 16);        // tensor_dim0 lo16
    g1[2] = (int)(((unsigned)KT & 0xFFFFu) << 16);        // dim0 hi=0 | dim1 lo16
    g1[3] = (int)(((unsigned)D_ & 0xFFFFu) << 16);        // dim1 hi=0 | tile_dim0
    g1[4] = (int)((unsigned)KT);                          // tile_dim1 | tile_dim2=0
    g1[5] = D_;                                           // tensor_dim0_stride lo32
    g1[6] = 0;
    g1[7] = 0;
    const i32x4 gz = {0, 0, 0, 0};

    auto tdm_load_k = [&](int kt0, int buf) {
        const unsigned long long ga =
            (unsigned long long)(const void*)(kp + (size_t)kt0 * D_);
        u32x4 g0;
        g0[0] = 1u;                                        // count=1
        g0[1] = (unsigned int)(unsigned long long)(void*)&Ksh[buf][0];
        g0[2] = (unsigned int)ga;                          // global_addr lo
        g0[3] = (unsigned int)(ga >> 32) | (2u << 30);     // ga hi | type=2
#if __clang_major__ >= 23
        const i32x8 g4 = {0, 0, 0, 0, 0, 0, 0, 0};
        __builtin_amdgcn_tensor_load_to_lds(g0, g1, gz, gz, g4, 0);
#else
        __builtin_amdgcn_tensor_load_to_lds(g0, g1, gz, gz, 0);
#endif
    };
#endif

    v8f ot[4];                         // Oᵀ accumulators (4 d-blocks of 16)
#pragma unroll
    for (int i = 0; i < 4; i++)
#pragma unroll
        for (int r = 0; r < 8; r++) ot[i][r] = 0.0f;

    float mrow = -1e30f;               // running max (log2-scaled domain)
    float lrow = 0.0f;                 // running sum (lane-local half-row)
    const float sl = 0.125f * 1.4426950408889634f;   // D^-0.5 * log2(e)

    // ---- Prologue: stage tile 0 into buffer 0 -----------------------------
#if USE_TDM
    if (wave == 0) tdm_load_k(0, 0);
#else
    {
        const uv4 kv = *(const uv4*)(kp + (size_t)kr * D_ + kc);
        *(uv4*)&Ksh[0][kr * KPITCH + kc] = kv;
    }
#endif
    {
        const uv4 vv = *(const uv4*)(vp + (size_t)kr * D_ + kc);
        const unsigned short* pv = (const unsigned short*)&vv;
#pragma unroll
        for (int j = 0; j < 8; j++)
            VTsh[0][(kc + j) * VPITCH + kr] = pv[j];
    }
#if USE_TDM
    if (wave == 0) __builtin_amdgcn_s_wait_tensorcnt(0);
#endif
    __syncthreads();

    // ---- Main loop: compute tile `it` while DMA'ing tile `it+1` -----------
    for (int it = 0; it < NT; ++it) {
        const int cur = it & 1;
        const int nxt = cur ^ 1;
        const bool has_next = (it + 1) < NT;
        const int kt1 = (it + 1) * KT;

        uv4 vnext;                                  // V prefetch (registers)
#if !USE_TDM
        uv4 knext;
#endif
        if (has_next) {
            vnext = *(const uv4*)(vp + (size_t)(kt1 + kr) * D_ + kc);
#if USE_TDM
            if (wave == 0) tdm_load_k(kt1, nxt);    // async DMA next K tile
#else
            knext = *(const uv4*)(kp + (size_t)(kt1 + kr) * D_ + kc);
#endif
        }

        // Sᵀ tiles: [32 keys x 16 qrows] as two 16x16 C-tiles
        v8f st[2];
#pragma unroll
        for (int t = 0; t < 2; t++) {
            v8f c;
#pragma unroll
            for (int r = 0; r < 8; r++) c[r] = 0.0f;
#pragma unroll
            for (int db = 0; db < 2; db++) {
                Frag a;
                const unsigned short* p =
                    &Ksh[cur][(t * 16 + ln) * KPITCH + db * 32 + 8 * hl];
                a.q4[0] = *(const uv4*)p;
                a.q4[1] = *(const uv4*)(p + 16);
                c = wmma_bf16(a, qf[db], c);
            }
            st[t] = c;
        }

        // Online softmax (transposed layout: row n lives in lanes n, n+16)
        float tmax = -1e30f;
#pragma unroll
        for (int t = 0; t < 2; t++)
#pragma unroll
            for (int r = 0; r < 8; r++) {
                st[t][r] *= sl;
                tmax = fmaxf(tmax, st[t][r]);
            }
        tmax = fmaxf(tmax, __shfl_xor(tmax, 16, 32));
        const float mnew = fmaxf(mrow, tmax);
        const float corr = exp2f(mrow - mnew);
        lrow *= corr;
#pragma unroll
        for (int i = 0; i < 4; i++)
#pragma unroll
            for (int r = 0; r < 8; r++) ot[i][r] *= corr;

        // P = exp2(st - mnew); in-lane repack Sᵀ C-tiles -> B-fragment
        Frag pf;
        float ls = 0.0f;
#pragma unroll
        for (int t = 0; t < 2; t++)
#pragma unroll
            for (int r = 0; r < 8; r++) {
                const float e = exp2f(st[t][r] - mnew);
                ls += e;
                pf.us[t * 8 + r] = f32_to_bf16(e);
            }
        lrow += ls;
        mrow  = mnew;

        // Oᵀ += Vᵀ (from LDS, pre-transposed) @ P
#pragma unroll
        for (int blk = 0; blk < 4; blk++) {
            Frag a;
            const unsigned short* p =
                &VTsh[cur][(blk * 16 + ln) * VPITCH + 8 * hl];
            a.q4[0] = *(const uv4*)p;
            a.q4[1] = *(const uv4*)(p + 16);
            ot[blk] = wmma_bf16(a, pf, ot[blk]);
        }

        // Drain prefetches into the next buffer (behind the WMMA work)
        if (has_next) {
#if !USE_TDM
            *(uv4*)&Ksh[nxt][kr * KPITCH + kc] = knext;
#endif
            const unsigned short* pv = (const unsigned short*)&vnext;
#pragma unroll
            for (int j = 0; j < 8; j++)
                VTsh[nxt][(kc + j) * VPITCH + kr] = pv[j];
        }
#if USE_TDM
        if (wave == 0) __builtin_amdgcn_s_wait_tensorcnt(0);
#endif
        __syncthreads();
    }

    // Finalize: combine half-row sums, normalize, store [B,N,768] bf16
    const float lt  = lrow + __shfl_xor(lrow, 16, 32);
    const float inv = 1.0f / lt;
    unsigned short* op =
        o + (size_t)(b * N_ + qrow + ln) * C_ + h * D_;
#pragma unroll
    for (int blk = 0; blk < 4; blk++) {
        union { unsigned short us[8]; uv4 v; } pk;
#pragma unroll
        for (int r = 0; r < 8; r++)
            pk.us[r] = f32_to_bf16(ot[blk][r] * inv);
        *(uv4*)(op + blk * 16 + 8 * hl) = pk.v;
    }
}

// ---------------------------------------------------------------------------
// Proj GEMM: [8192,768] x [768,768]^T + bias -> f32 output [B,N,C]
// ---------------------------------------------------------------------------
__global__ __launch_bounds__(256) void proj_gemm_kernel(
    const unsigned short* __restrict__ ab,   // attn out bf16 [8192,768]
    const unsigned short* __restrict__ wb,   // proj_w bf16 [768,768]
    const float*          __restrict__ bias, // [768]
    float* __restrict__ out)                 // [8192,768] f32
{
    const int wave  = threadIdx.x >> 5;
    const int lane  = threadIdx.x & 31;
    const int ln    = lane & 15;
    const int hl    = lane >> 4;
    const int Mbase = blockIdx.y * 128 + (wave & 3) * 32;
    const int Obase = blockIdx.x * 128 + (wave >> 2) * 64;

    v8f acc[2][4];
#pragma unroll
    for (int i = 0; i < 2; i++)
#pragma unroll
        for (int j = 0; j < 4; j++)
#pragma unroll
            for (int r = 0; r < 8; r++) acc[i][j][r] = 0.0f;

#pragma unroll 2
    for (int k0 = 0; k0 < C_; k0 += 32) {
        Frag a[2];
#pragma unroll
        for (int mm = 0; mm < 2; mm++) {
            const unsigned short* p =
                ab + (size_t)(Mbase + mm * 16 + ln) * C_ + k0 + 8 * hl;
            a[mm].q4[0] = *(const uv4*)p;
            a[mm].q4[1] = *(const uv4*)(p + 16);
        }
#pragma unroll
        for (int nn = 0; nn < 4; nn++) {
            Frag bf;
            const unsigned short* p =
                wb + (size_t)(Obase + nn * 16 + ln) * C_ + k0 + 8 * hl;
            bf.q4[0] = *(const uv4*)p;
            bf.q4[1] = *(const uv4*)(p + 16);
#pragma unroll
            for (int mm = 0; mm < 2; mm++)
                acc[mm][nn] = wmma_bf16(a[mm], bf, acc[mm][nn]);
        }
    }

#pragma unroll
    for (int nn = 0; nn < 4; nn++) {
        const int o = Obase + nn * 16 + ln;
        const float bv = bias[o];
#pragma unroll
        for (int mm = 0; mm < 2; mm++) {
            const int mrow = Mbase + mm * 16 + 8 * hl;
#pragma unroll
            for (int r = 0; r < 8; r++)
                out[(size_t)(mrow + r) * C_ + o] = acc[mm][nn][r] + bv;
        }
    }
}

// ---------------------------------------------------------------------------
// Host launch
// ---------------------------------------------------------------------------
extern "C" void kernel_launch(void* const* d_in, const int* in_sizes, int n_in,
                              void* d_out, int out_size, void* d_ws, size_t ws_size,
                              hipStream_t stream) {
    (void)in_sizes; (void)n_in; (void)out_size; (void)ws_size;

    const float* x      = (const float*)d_in[0];   // [4,2048,768]
    const float* qkv_w  = (const float*)d_in[1];   // [2304,768]
    const float* qkv_b  = (const float*)d_in[2];   // [2304]
    const float* proj_w = (const float*)d_in[3];   // [768,768]
    const float* proj_b = (const float*)d_in[4];   // [768]
    float* out = (float*)d_out;                    // [4,2048,768]

    char* ws = (char*)d_ws;
    size_t off = 0;
    auto carve = [&](size_t elems) -> unsigned short* {
        unsigned short* p = (unsigned short*)(ws + off);
        off += ((elems * 2) + 255) & ~(size_t)255;
        return p;
    };
    unsigned short* xb    = carve((size_t)M_ * C_);        // x bf16
    unsigned short* wqkv  = carve((size_t)OC3 * C_);       // qkv_w bf16
    unsigned short* wproj = carve((size_t)C_ * C_);        // proj_w bf16
    unsigned short* qb    = carve((size_t)B_ * H_ * N_ * D_);
    unsigned short* kb    = carve((size_t)B_ * H_ * N_ * D_);
    unsigned short* vb    = carve((size_t)B_ * H_ * N_ * D_);
    unsigned short* ao    = carve((size_t)M_ * C_);        // attn out bf16

    const int nx = M_ * C_, nw1 = OC3 * C_, nw2 = C_ * C_;
    cvt_f32_bf16_kernel<<<dim3((nx  + 255) / 256 < 2048 ? (nx  + 255) / 256 : 2048), 256, 0, stream>>>(x,      xb,    nx);
    cvt_f32_bf16_kernel<<<dim3((nw1 + 255) / 256 < 2048 ? (nw1 + 255) / 256 : 2048), 256, 0, stream>>>(qkv_w,  wqkv,  nw1);
    cvt_f32_bf16_kernel<<<dim3((nw2 + 255) / 256 < 2048 ? (nw2 + 255) / 256 : 2048), 256, 0, stream>>>(proj_w, wproj, nw2);

    qkv_gemm_kernel<<<dim3(OC3 / 128, M_ / 128), 256, 0, stream>>>(
        xb, wqkv, qkv_b, qb, kb, vb);

    flash_attn_kernel<<<dim3(N_ / 128, H_, B_), 256, 0, stream>>>(
        qb, kb, vb, ao);

    proj_gemm_kernel<<<dim3(C_ / 128, M_ / 128), 256, 0, stream>>>(
        ao, wproj, proj_b, out);
}
